// view_learner_9259949490648
// MI455X (gfx1250) — matrix operand order repaired and checked by reading, not compiled, not run on hardware
//
#include <hip/hip_runtime.h>

// ---------------------------------------------------------------------------
// view_learner additive branch for MI455X (gfx1250, wave32, WMMA)
//   x_t = x + exp(0.5*logvar)*noise                        (elementwise)
//   S[i] = sum_j exp(-0.5*max(||xi||^2+||xj||^2-2 xi.xj,0)/var)   (fused GEMM)
//   kl   = (log B - mean_i log S[i]) / ln 2
// Gram matrix via split-bf16 (hi+lo, 3 WMMA products) + exact diagonal.
// Symmetry: only upper-triangular 128x128 tiles; off-diagonal tiles feed both
// row-sums (rows of bi) and column-sums (rows of bj) -> halves WMMA work.
// Staging via async global->LDS (ASYNCcnt) with double-buffered LDS tiles.
// ---------------------------------------------------------------------------

#define B_N  8192
#define K_F  3072
#define TILE 128
#define KB   32
#define RS   40   // LDS row stride in bf16 elements (32 data + 8 pad)
#define NBLK (B_N / TILE)                   // 64
#define NTRI (NBLK * (NBLK + 1) / 2)        // 2080

typedef __attribute__((ext_vector_type(16))) __bf16 v16bf;
typedef __attribute__((ext_vector_type(8)))  __bf16 v8bf;
typedef __attribute__((ext_vector_type(8)))  float  v8f;

union V16 { v16bf v; v8bf h[2]; };

// pointer types matching __builtin_amdgcn_global_load_async_to_lds_b128
typedef int vi4 __attribute__((vector_size(16)));
typedef __attribute__((address_space(1))) vi4* gp4;   // global src
typedef __attribute__((address_space(3))) vi4* lp4;   // LDS dst

// async 16B global -> LDS copy (per-lane addresses), tracked by ASYNCcnt
#define ASYNC_CP16(g, l)                                                 \
    __builtin_amdgcn_global_load_async_to_lds_b128((gp4)(g), (lp4)(l), 0, 0)

// workspace layout (bytes)
static const size_t OFF_XHI = 0;                                   // 50331648 B
static const size_t OFF_XLO = (size_t)B_N * K_F * 2;               // 50331648 B
static const size_t OFF_XN  = OFF_XLO + (size_t)B_N * K_F * 2;     // 32768 B
static const size_t OFF_S   = OFF_XN + (size_t)B_N * 4;            // 32768 B

__device__ __forceinline__ unsigned short f2bf_rne(float f) {
    unsigned u = __float_as_uint(f);
    u += 0x7FFFu + ((u >> 16) & 1u);          // round-to-nearest-even
    return (unsigned short)(u >> 16);
}

__device__ __forceinline__ void split_bf16(float f, unsigned short& hi, unsigned short& lo) {
    hi = f2bf_rne(f);
    float hf = __uint_as_float(((unsigned)hi) << 16);
    lo = f2bf_rne(f - hf);
}

// ---------------------------------------------------------------------------
// Kernel 1: per row — x_t, bf16 hi/lo split, exact fp32 row norm, zero S[row]
// ---------------------------------------------------------------------------
__global__ __launch_bounds__(256) void prep_kernel(
    const float* __restrict__ x, const float* __restrict__ noise,
    const float* __restrict__ logvar, float* __restrict__ xt,
    unsigned short* __restrict__ Xhi, unsigned short* __restrict__ Xlo,
    float* __restrict__ xn, float* __restrict__ S)
{
    __shared__ float red[256];
    const int row = blockIdx.x;
    const int t   = threadIdx.x;
    const float s = __expf(0.5f * logvar[0]);

    const float4* x4 = (const float4*)(x     + (size_t)row * K_F);
    const float4* n4 = (const float4*)(noise + (size_t)row * K_F);
    float4*       o4 = (float4*)(xt + (size_t)row * K_F);
    ushort4*      h4 = (ushort4*)(Xhi + (size_t)row * K_F);
    ushort4*      l4 = (ushort4*)(Xlo + (size_t)row * K_F);

    float nrm = 0.f;
#pragma unroll
    for (int v = 0; v < 3; ++v) {
        const int i = t + v * 256;                 // 768 float4 per row
        float4 xv = x4[i];
        float4 nv = n4[i];
        float4 ov;
        ov.x = fmaf(s, nv.x, xv.x); ov.y = fmaf(s, nv.y, xv.y);
        ov.z = fmaf(s, nv.z, xv.z); ov.w = fmaf(s, nv.w, xv.w);
        o4[i] = ov;
        ushort4 hv, lv;
        split_bf16(xv.x, hv.x, lv.x); split_bf16(xv.y, hv.y, lv.y);
        split_bf16(xv.z, hv.z, lv.z); split_bf16(xv.w, hv.w, lv.w);
        h4[i] = hv; l4[i] = lv;
        nrm += xv.x*xv.x + xv.y*xv.y + xv.z*xv.z + xv.w*xv.w;
    }
    red[t] = nrm;
    __syncthreads();
    for (int off = 128; off; off >>= 1) {
        if (t < off) red[t] += red[t + off];
        __syncthreads();
    }
    if (t == 0) { xn[row] = red[0]; S[row] = 0.f; }   // reset accumulator each launch
}

// ---------------------------------------------------------------------------
// Kernel 2: upper-triangular 128x128 Gram tiles, split-bf16 WMMA,
//           double-buffered async staging, fused exp + row/col sums.
// ---------------------------------------------------------------------------
__global__ __launch_bounds__(256) void gram_lse_kernel(
    const unsigned short* __restrict__ Xhi, const unsigned short* __restrict__ Xlo,
    const float* __restrict__ xn, const float* __restrict__ logvar,
    float* __restrict__ S)
{
    // [buffer][array: 0=Ahi 1=Alo 2=Bhi 3=Blo][tile]
    __shared__ __align__(16) unsigned short tiles[2][4][TILE * RS];
    __shared__ float xnI[TILE], xnJ[TILE];

    // triangular decode: tile (a,b) with a<=b, linear index blockIdx.x
    int p = blockIdx.x, a = 0;
    while (p >= NBLK - a) { p -= NBLK - a; ++a; }
    const int bi = a * TILE;              // row block
    const int bj = (a + p) * TILE;        // col block (>= bi)

    const int t    = threadIdx.x;
    const int w    = t >> 5;              // 8 waves
    const int lane = t & 31;
    const int waveM = (w & 3) * 32;       // 4 wave-rows of 32 M
    const int waveN = (w >> 2) * 64;      // 2 wave-cols of 64 N
    const int mrow  = lane & 15;

    if (t < 128) xnI[t] = xn[bi + t];
    else         xnJ[t - 128] = xn[bj + (t - 128)];

    v8f acc[2][4];
#pragma unroll
    for (int mt = 0; mt < 2; ++mt)
#pragma unroll
        for (int nt = 0; nt < 4; ++nt)
#pragma unroll
            for (int e = 0; e < 8; ++e) acc[mt][nt][e] = 0.f;

    // staging coords: thread -> (row r2, 16-element half of a 32-wide K chunk)
    const int    r2    = t >> 1;
    const int    halfk = t & 1;
    const size_t srcA  = (size_t)(bi + r2) * K_F + halfk * 16;
    const size_t srcB  = (size_t)(bj + r2) * K_F + halfk * 16;
    const int    ldst  = r2 * RS + halfk * 16;

    const int nIters = K_F / KB;          // 96

    // prologue: fill buffer 0
    {
        ASYNC_CP16(Xhi + srcA,     tiles[0][0] + ldst);
        ASYNC_CP16(Xhi + srcA + 8, tiles[0][0] + ldst + 8);
        ASYNC_CP16(Xlo + srcA,     tiles[0][1] + ldst);
        ASYNC_CP16(Xlo + srcA + 8, tiles[0][1] + ldst + 8);
        ASYNC_CP16(Xhi + srcB,     tiles[0][2] + ldst);
        ASYNC_CP16(Xhi + srcB + 8, tiles[0][2] + ldst + 8);
        ASYNC_CP16(Xlo + srcB,     tiles[0][3] + ldst);
        ASYNC_CP16(Xlo + srcB + 8, tiles[0][3] + ldst + 8);
    }

    for (int it = 0; it < nIters; ++it) {
        const int cur = it & 1;
        if (it + 1 < nIters) {
            // issue next chunk into the spare buffer (last read 2 barriers ago)
            const int kn = (it + 1) * KB;
            unsigned short* nAhi = tiles[cur ^ 1][0];
            unsigned short* nAlo = tiles[cur ^ 1][1];
            unsigned short* nBhi = tiles[cur ^ 1][2];
            unsigned short* nBlo = tiles[cur ^ 1][3];
            ASYNC_CP16(Xhi + srcA + kn,     nAhi + ldst);
            ASYNC_CP16(Xhi + srcA + kn + 8, nAhi + ldst + 8);
            ASYNC_CP16(Xlo + srcA + kn,     nAlo + ldst);
            ASYNC_CP16(Xlo + srcA + kn + 8, nAlo + ldst + 8);
            ASYNC_CP16(Xhi + srcB + kn,     nBhi + ldst);
            ASYNC_CP16(Xhi + srcB + kn + 8, nBhi + ldst + 8);
            ASYNC_CP16(Xlo + srcB + kn,     nBlo + ldst);
            ASYNC_CP16(Xlo + srcB + kn + 8, nBlo + ldst + 8);
            // async loads complete in order: <=8 outstanding => current buffer landed
            asm volatile("s_wait_asynccnt 0x8" ::: "memory");
        } else {
            asm volatile("s_wait_asynccnt 0x0" ::: "memory");
        }
        __syncthreads();   // all waves' async copies for 'cur' are visible

        const unsigned short* Ahi = tiles[cur][0];
        const unsigned short* Alo = tiles[cur][1];
        const unsigned short* Bhi = tiles[cur][2];
        const unsigned short* Blo = tiles[cur][3];

        // B fragments: lanes 0-15 hold K=0..15, lanes 16-31 hold K=16..31 (contig)
        V16 bh[4], bl[4];
        const int kb0 = (lane & 16) ? 16 : 0;
#pragma unroll
        for (int nt = 0; nt < 4; ++nt) {
            const int n = waveN + nt * 16 + mrow;
            const v8bf* pb = (const v8bf*)(Bhi + n * RS + kb0);
            bh[nt].h[0] = pb[0]; bh[nt].h[1] = pb[1];
            const v8bf* pl = (const v8bf*)(Blo + n * RS + kb0);
            bl[nt].h[0] = pl[0]; bl[nt].h[1] = pl[1];
        }
        // A fragments: lanes 0-15 hold K=0..15, lanes 16-31 hold K=16..31 (contig)
        const int ka0 = (lane & 16) ? 16 : 0;
#pragma unroll
        for (int mt = 0; mt < 2; ++mt) {
            const int m = waveM + mt * 16 + mrow;
            V16 ah, al;
            const v8bf* pa = (const v8bf*)(Ahi + m * RS + ka0);
            ah.h[0] = pa[0]; ah.h[1] = pa[1];
            const v8bf* pl = (const v8bf*)(Alo + m * RS + ka0);
            al.h[0] = pl[0]; al.h[1] = pl[1];
#pragma unroll
            for (int nt = 0; nt < 4; ++nt) {
                acc[mt][nt] = __builtin_amdgcn_wmma_f32_16x16x32_bf16(
                    false, ah.v, false, bh[nt].v, (short)0, acc[mt][nt], false, false);
                acc[mt][nt] = __builtin_amdgcn_wmma_f32_16x16x32_bf16(
                    false, ah.v, false, bl[nt].v, (short)0, acc[mt][nt], false, false);
                acc[mt][nt] = __builtin_amdgcn_wmma_f32_16x16x32_bf16(
                    false, al.v, false, bh[nt].v, (short)0, acc[mt][nt], false, false);
            }
        }
        __syncthreads();   // protect spare buffer before next iteration's issue
    }

    // epilogue: dist -> exp -> row sums (always) + column sums (bi != bj).
    // All exponents <= 0; diagonal term is exactly 1.
    const float var = __expf(logvar[0]) + 1e-10f;
    const float c   = -0.72134752044f / var;       // -0.5 / (var * ln2)
    const int rhalf = (lane & 16) ? 8 : 0;

    float csum[4] = {0.f, 0.f, 0.f, 0.f};          // per-lane column partials

#pragma unroll
    for (int mt = 0; mt < 2; ++mt) {
        float rowsum[8] = {0.f, 0.f, 0.f, 0.f, 0.f, 0.f, 0.f, 0.f};
#pragma unroll
        for (int nt = 0; nt < 4; ++nt) {
            const int col_l = waveN + nt * 16 + mrow;
            const int gcol  = bj + col_l;
            const float xj  = xnJ[col_l];
#pragma unroll
            for (int r = 0; r < 8; ++r) {
                const int row_l = waveM + mt * 16 + rhalf + r;
                float dist = xnI[row_l] + xj - 2.f * acc[mt][nt][r];
                dist = fmaxf(dist, 0.f);
                const float term = (bi + row_l == gcol)
                                   ? 1.f
                                   : __builtin_amdgcn_exp2f(dist * c);
                rowsum[r] += term;
                csum[nt]  += term;
            }
        }
#pragma unroll
        for (int r = 0; r < 8; ++r) {
            float v = rowsum[r];
            v += __shfl_xor(v, 1, 16);
            v += __shfl_xor(v, 2, 16);
            v += __shfl_xor(v, 4, 16);
            v += __shfl_xor(v, 8, 16);
            if (mrow == 0)                          // lanes 0 and 16
                atomicAdd(&S[bi + waveM + mt * 16 + rhalf + r], v);
        }
    }

    if (bi != bj) {
        // symmetry: column j's partial (over this wave's 32 rows) -> S[j]
        // column c lives in lanes c (M 0-7) and c+16 (M 8-15): one xor-16 combine
#pragma unroll
        for (int nt = 0; nt < 4; ++nt) {
            float v = csum[nt];
            v += __shfl_xor(v, 16, 32);
            if (lane < 16)
                atomicAdd(&S[bj + waveN + nt * 16 + mrow], v);
        }
    }
}

// ---------------------------------------------------------------------------
// Kernel 3: kl = (log B - mean_i log S[i]) / ln2   -> d_out[B*K_F]
// ---------------------------------------------------------------------------
__global__ __launch_bounds__(256) void finalize_kernel(
    const float* __restrict__ S, float* __restrict__ out)
{
    __shared__ float red[256];
    const int t = threadIdx.x;
    float acc = 0.f;
    for (int i = t; i < B_N; i += 256) acc += __logf(S[i]);
    red[t] = acc;
    __syncthreads();
    for (int off = 128; off; off >>= 1) {
        if (t < off) red[t] += red[t + off];
        __syncthreads();
    }
    if (t == 0)
        out[(size_t)B_N * K_F] =
            (logf((float)B_N) - red[0] / (float)B_N) * 1.44269504089f;
}

extern "C" void kernel_launch(void* const* d_in, const int* in_sizes, int n_in,
                              void* d_out, int out_size, void* d_ws, size_t ws_size,
                              hipStream_t stream) {
    (void)in_sizes; (void)n_in; (void)out_size; (void)ws_size;
    const float* x      = (const float*)d_in[0];
    const float* noise  = (const float*)d_in[1];
    const float* logvar = (const float*)d_in[2];
    float* out = (float*)d_out;

    char* ws = (char*)d_ws;
    unsigned short* Xhi = (unsigned short*)(ws + OFF_XHI);
    unsigned short* Xlo = (unsigned short*)(ws + OFF_XLO);
    float* xn = (float*)(ws + OFF_XN);
    float* S  = (float*)(ws + OFF_S);

    prep_kernel<<<B_N, 256, 0, stream>>>(x, noise, logvar, out, Xhi, Xlo, xn, S);
    gram_lse_kernel<<<NTRI, 256, 0, stream>>>(Xhi, Xlo, xn, logvar, S);
    finalize_kernel<<<1, 256, 0, stream>>>(S, out);
}
